// KalmanNet_8392366096686
// MI455X (gfx1250) — matrix operand order repaired and checked by reading, not compiled, not run on hardware
//
#include <hip/hip_runtime.h>

// ---------------------------------------------------------------------------
// KalmanNet (GRU Kalman filter) for MI455X / gfx1250.
// BATCH=1024 sequences, SEQ=512 sequential steps, HIDDEN=256, gates=768.
// Persistent workgroups: each WG owns BT=32 batch rows for the whole sequence.
// Per step: two [32,256]x[256,768] GEMMs via v_wmma_f32_16x16x32_bf16.
// GEMM loops process both 16-row M-tiles per N-tile: weights (B fragments)
// loaded once per N-tile and shared -> 2x less L2 traffic, 2 independent
// WMMA accumulator chains per wave for ILP.
// Weights converted+swizzled to bf16 B-fragment layout once per launch (d_ws).
// ~184KB dynamic LDS per WG (CDNA5: 320KB/WGP).
// ---------------------------------------------------------------------------

#define BT      32      // batch rows per workgroup
#define HID     256
#define G3      768     // 3*HID
#define SD      4       // state dim
#define OD      2       // obs dim
#define SEQLEN  512
#define BATCHN  1024
#define NT_CNT  48      // 768/16 N-tiles
#define KT_CNT  8       // 256/32 K-steps
#define NTHREADS 256

typedef __bf16 bf16_t;
typedef __bf16 v16bf __attribute__((ext_vector_type(16)));
typedef float  v8f   __attribute__((ext_vector_type(8)));

// ---- dynamic LDS layout (bytes) -------------------------------------------
#define SMEM_UB    0        // bf16 u fragments  [2][8][32][16]      16384
#define SMEM_HB    16384    // bf16 h fragments  [2][8][32][16]      16384
#define SMEM_G     32768    // f32 g  [BT][768]                      98304
#define SMEM_HN    131072   // f32 hn [BT][256]  (h_n, then h_new)   32768
#define SMEM_WIN   163840   // f32 W_in [256][6]                      6144
#define SMEM_WOUT  169984   // f32 W_out [8][256]                     8192
#define SMEM_BIN   178176   // f32 b_in [256]                         1024
#define SMEM_BIH   179200   // f32 b_ih [768]                         3072
#define SMEM_BHH   182272   // f32 b_hh [768]                         3072
#define SMEM_BOUT  185344   // f32 b_out [8] (pad 64)                   64
#define SMEM_FM    185408   // f32 F [4][4]                             64
#define SMEM_HM    185472   // f32 H [2][4] (pad 64)                    64
#define SMEM_XH    185536   // f32 x_hat  [BT][4]                      512
#define SMEM_XP    186048   // f32 x_prev [BT][4]                      512
#define SMEM_XPRI  186560   // f32 x_pri  [BT][4]                      512
#define SMEM_INN   187072   // f32 innov  [BT][2]                      256
#define SMEM_KB    187328   // f32 K gains [BT][8]                    1024
#define SMEM_TOTAL 188352

// A-fragment swizzle: 16-bit A matrix 16x32 (ISA 7.12.2). Element (row r of
// the 32-row batch tile, feature j of 256) -> flat index into [2][8][32][16].
__device__ __forceinline__ int swizA(int r, int j) {
  int mt = r >> 4, M = r & 15;
  int kt = j >> 5, kk = j & 31;
  int khalf = (kk >> 3) & 1;              // lanes 16-31 hold K%16 in [8,16)
  int e     = (kk & 7) + ((kk >> 4) << 3);// V4-7 hold K>=16
  int lanev = (khalf << 4) | M;
  return ((mt * KT_CNT + kt) * 32 + lanev) * 16 + e;
}

__device__ __forceinline__ float sigm_f(float x) {
  return 1.f / (1.f + __expf(-x));
}
__device__ __forceinline__ float tanh_f(float x) {
  x = fminf(15.f, fmaxf(-15.f, x));
  float e = __expf(2.f * x);
  return (e - 1.f) / (e + 1.f);
}

// ---------------------------------------------------------------------------
// Prep: convert W_ih / W_hh fp32 [768][256] -> bf16 in B-fragment order:
//   out[((g*48+nt)*8+kt)*32*16 + lane*16 + e],
//   lane: n = lane&15, khalf = lane>>4 ; element e: K = kt*32 + khalf*8 + (e&7) + (e>>3)*16.
// One contiguous 32B load per lane per fragment at GEMM time.
// ---------------------------------------------------------------------------
__global__ void knet_prep(const float* __restrict__ Wih,
                          const float* __restrict__ Whh,
                          bf16_t* __restrict__ out) {
  int idx = blockIdx.x * NTHREADS + threadIdx.x;
  if (idx >= 2 * G3 * HID) return;
  int e = idx & 15;
  int t = idx >> 4;
  int lanev = t & 31; t >>= 5;
  int kt = t & 7;     t >>= 3;
  int nt = t % NT_CNT;
  int g  = t / NT_CNT;
  int n = lanev & 15, khalf = lanev >> 4;
  int ng = nt * 16 + n;
  int kg = kt * 32 + khalf * 8 + (e & 7) + ((e >> 3) << 4);
  const float* W = g ? Whh : Wih;
  out[idx] = (bf16_t)W[ng * HID + kg];
}

// ---------------------------------------------------------------------------
// Main persistent kernel: grid = BATCHN/BT workgroups of 256 threads (8 waves)
// ---------------------------------------------------------------------------
__global__ __launch_bounds__(NTHREADS, 1)
void knet_main(const float* __restrict__ y_seq,
               const float* __restrict__ Fm,
               const float* __restrict__ Hm,
               const float* __restrict__ W_in,
               const float* __restrict__ b_in,
               const float* __restrict__ b_ih,
               const float* __restrict__ b_hh,
               const float* __restrict__ W_out,
               const float* __restrict__ b_out,
               const bf16_t* __restrict__ wW,   // swizzled bf16 W_ih|W_hh
               float* __restrict__ out) {
  extern __shared__ __align__(32) char smem[];
  bf16_t* s_uB   = (bf16_t*)(smem + SMEM_UB);
  bf16_t* s_hB   = (bf16_t*)(smem + SMEM_HB);
  float*  s_g    = (float*)(smem + SMEM_G);
  float*  s_hn   = (float*)(smem + SMEM_HN);
  float*  s_Win  = (float*)(smem + SMEM_WIN);
  float*  s_Wout = (float*)(smem + SMEM_WOUT);
  float*  s_bin  = (float*)(smem + SMEM_BIN);
  float*  s_bih  = (float*)(smem + SMEM_BIH);
  float*  s_bhh  = (float*)(smem + SMEM_BHH);
  float*  s_bout = (float*)(smem + SMEM_BOUT);
  float*  s_F    = (float*)(smem + SMEM_FM);
  float*  s_H    = (float*)(smem + SMEM_HM);
  float*  s_xh   = (float*)(smem + SMEM_XH);
  float*  s_xp   = (float*)(smem + SMEM_XP);
  float*  s_xpri = (float*)(smem + SMEM_XPRI);
  float*  s_inn  = (float*)(smem + SMEM_INN);
  float*  s_K    = (float*)(smem + SMEM_KB);

  const int tid  = threadIdx.x;
  const int lane = tid & 31;
  const int wave = tid >> 5;
  const int row0 = blockIdx.x * BT;

  // ---- init: stage small weights/biases to LDS, zero state ---------------
  for (int i = tid; i < HID * 6; i += NTHREADS) s_Win[i]  = W_in[i];
  for (int i = tid; i < 8 * HID; i += NTHREADS) s_Wout[i] = W_out[i];
  for (int i = tid; i < HID;     i += NTHREADS) s_bin[i]  = b_in[i];
  for (int i = tid; i < G3;      i += NTHREADS) { s_bih[i] = b_ih[i]; s_bhh[i] = b_hh[i]; }
  if (tid < 16) s_F[tid] = Fm[tid];
  if (tid < 8)  { s_H[tid] = Hm[tid]; s_bout[tid] = b_out[tid]; }
  for (int i = tid; i < BT * SD; i += NTHREADS) { s_xh[i] = 0.f; s_xp[i] = 0.f; }
  for (int i = tid; i < 2 * KT_CNT * 32 * 16; i += NTHREADS) s_hB[i] = (bf16_t)0.f;
  __syncthreads();

  for (int t = 0; t < SEQLEN; ++t) {
    // ---- A1: Kalman predict + innovation (one thread per row) ------------
    if (tid < BT) {
      int r = tid;
      float x0 = s_xh[r * 4 + 0], x1 = s_xh[r * 4 + 1];
      float x2 = s_xh[r * 4 + 2], x3 = s_xh[r * 4 + 3];
      float xp[SD];
      #pragma unroll
      for (int s = 0; s < SD; ++s) {
        xp[s] = s_F[s * 4 + 0] * x0 + s_F[s * 4 + 1] * x1 +
                s_F[s * 4 + 2] * x2 + s_F[s * 4 + 3] * x3;
        s_xpri[r * 4 + s] = xp[s];
      }
      const float* yp = y_seq + ((size_t)(row0 + r) * SEQLEN + t) * OD;
      #pragma unroll
      for (int o = 0; o < OD; ++o) {
        float hx = s_H[o * 4 + 0] * xp[0] + s_H[o * 4 + 1] * xp[1] +
                   s_H[o * 4 + 2] * xp[2] + s_H[o * 4 + 3] * xp[3];
        s_inn[r * 2 + o] = yp[o] - hx;
      }
    }
    __syncthreads();

    // ---- A2: u = W_in @ [dx, innov] + b_in  -> bf16 A-fragments ----------
    {
      int j = tid;                      // feature 0..255
      const float* wj = s_Win + j * 6;
      float bj = s_bin[j];
      for (int r = 0; r < BT; ++r) {
        float d0 = s_xh[r * 4 + 0] - s_xp[r * 4 + 0];
        float d1 = s_xh[r * 4 + 1] - s_xp[r * 4 + 1];
        float d2 = s_xh[r * 4 + 2] - s_xp[r * 4 + 2];
        float d3 = s_xh[r * 4 + 3] - s_xp[r * 4 + 3];
        float u = bj + wj[0] * d0 + wj[1] * d1 + wj[2] * d2 + wj[3] * d3 +
                  wj[4] * s_inn[r * 2 + 0] + wj[5] * s_inn[r * 2 + 1];
        s_uB[swizA(r, j)] = (bf16_t)u;
      }
    }
    __syncthreads();

    // ---- B: GEMM1  g = b_ih + u @ W_ih^T  (WMMA bf16) --------------------
    // Both M-tiles per N-tile: B fragments loaded once, 2 indep acc chains.
    for (int nt = wave; nt < NT_CNT; nt += 8) {
      int ncol = nt * 16 + (lane & 15);
      float binit = s_bih[ncol];
      v8f acc0, acc1;
      #pragma unroll
      for (int i = 0; i < 8; ++i) { acc0[i] = binit; acc1[i] = binit; }
      const bf16_t* Bp = wW + ((size_t)nt * KT_CNT) * 32 * 16;
      if (nt + 8 < NT_CNT)   // keep next N-tile's weights warming in cache
        __builtin_prefetch(wW + ((size_t)(nt + 8) * KT_CNT) * 32 * 16 + lane * 16, 0, 1);
      #pragma unroll
      for (int kt = 0; kt < KT_CNT; ++kt) {
        v16bf b  = *(const v16bf*)(Bp + (kt * 32 + lane) * 16);
        v16bf a0 = *(const v16bf*)(s_uB + ((0 * KT_CNT + kt) * 32 + lane) * 16);
        v16bf a1 = *(const v16bf*)(s_uB + ((1 * KT_CNT + kt) * 32 + lane) * 16);
        acc0 = __builtin_amdgcn_wmma_f32_16x16x32_bf16(
            false, a0, false, b, (short)0, acc0, false, false);
        acc1 = __builtin_amdgcn_wmma_f32_16x16x32_bf16(
            false, a1, false, b, (short)0, acc1, false, false);
      }
      int rowb = (lane >> 4) << 3;               // M = i + 8*(lane/16)
      float* gp0 = s_g + rowb * G3 + ncol;       // mt = 0
      float* gp1 = s_g + (16 + rowb) * G3 + ncol;// mt = 1
      #pragma unroll
      for (int i = 0; i < 8; ++i) { gp0[i * G3] = acc0[i]; gp1[i * G3] = acc1[i]; }
    }
    __syncthreads();

    // ---- C: GEMM2  gh = b_hh + h @ W_hh^T : r,z add into g; n -> s_hn ----
    for (int nt = wave; nt < NT_CNT; nt += 8) {
      int ncol = nt * 16 + (lane & 15);
      float binit = s_bhh[ncol];
      v8f acc0, acc1;
      #pragma unroll
      for (int i = 0; i < 8; ++i) { acc0[i] = binit; acc1[i] = binit; }
      const bf16_t* Bp = wW + ((size_t)(NT_CNT + nt) * KT_CNT) * 32 * 16;
      if (nt + 8 < NT_CNT)
        __builtin_prefetch(wW + ((size_t)(NT_CNT + nt + 8) * KT_CNT) * 32 * 16 + lane * 16, 0, 1);
      #pragma unroll
      for (int kt = 0; kt < KT_CNT; ++kt) {
        v16bf b  = *(const v16bf*)(Bp + (kt * 32 + lane) * 16);
        v16bf a0 = *(const v16bf*)(s_hB + ((0 * KT_CNT + kt) * 32 + lane) * 16);
        v16bf a1 = *(const v16bf*)(s_hB + ((1 * KT_CNT + kt) * 32 + lane) * 16);
        acc0 = __builtin_amdgcn_wmma_f32_16x16x32_bf16(
            false, a0, false, b, (short)0, acc0, false, false);
        acc1 = __builtin_amdgcn_wmma_f32_16x16x32_bf16(
            false, a1, false, b, (short)0, acc1, false, false);
      }
      int rowb = (lane >> 4) << 3;
      if (nt < 32) {                        // r,z gates: g += gh
        float* gp0 = s_g + rowb * G3 + ncol;
        float* gp1 = s_g + (16 + rowb) * G3 + ncol;
        #pragma unroll
        for (int i = 0; i < 8; ++i) { gp0[i * G3] += acc0[i]; gp1[i * G3] += acc1[i]; }
      } else {                              // n gate h-side kept separate
        int nc = ncol - 2 * HID;
        float* hp0 = s_hn + rowb * HID + nc;
        float* hp1 = s_hn + (16 + rowb) * HID + nc;
        #pragma unroll
        for (int i = 0; i < 8; ++i) { hp0[i * HID] = acc0[i]; hp1[i * HID] = acc1[i]; }
      }
    }
    __syncthreads();

    // ---- D: GRU gates, h update (torch order r,z,n) ----------------------
    {
      int j = tid;
      for (int r = 0; r < BT; ++r) {
        float gr = s_g[r * G3 + j];
        float gz = s_g[r * G3 + HID + j];
        float gn = s_g[r * G3 + 2 * HID + j];   // i_n (incl. b_ih)
        float hnv = s_hn[r * HID + j];          // h_n (incl. b_hh)
        float rg = sigm_f(gr);
        float z  = sigm_f(gz);
        float n  = tanh_f(gn + rg * hnv);
        int sa = swizA(r, j);
        float hold = (float)s_hB[sa];
        float hnew = (1.f - z) * n + z * hold;
        s_hB[sa] = (bf16_t)hnew;                // bf16 A-fragment for next step
        s_hn[r * HID + j] = hnew;               // f32 h_new for W_out
      }
    }
    __syncthreads();

    // ---- E: K = W_out @ h_new + b_out  (one thread per (row, out)) -------
    {
      int r = tid >> 3, o = tid & 7;
      const float* wo = s_Wout + o * HID;
      const float* hv = s_hn + r * HID;
      float acc = s_bout[o];
      #pragma unroll 4
      for (int k = 0; k < HID; ++k) acc = fmaf(wo[k], hv[k], acc);
      s_K[r * 8 + o] = acc;
    }
    __syncthreads();

    // ---- F: x_new = x_pri + K @ innov ; roll state; emit output ----------
    if (tid < BT * SD) {
      int r = tid >> 2, s = tid & 3;
      float xn = s_xpri[r * 4 + s] +
                 s_K[r * 8 + s * 2 + 0] * s_inn[r * 2 + 0] +
                 s_K[r * 8 + s * 2 + 1] * s_inn[r * 2 + 1];
      out[((size_t)(row0 + r) * SEQLEN + t) * SD + s] = xn;
      s_xp[r * 4 + s] = s_xh[r * 4 + s];
      s_xh[r * 4 + s] = xn;
    }
    __syncthreads();
  }
}

// ---------------------------------------------------------------------------
extern "C" void kernel_launch(void* const* d_in, const int* in_sizes, int n_in,
                              void* d_out, int out_size, void* d_ws, size_t ws_size,
                              hipStream_t stream) {
  const float* y_seq = (const float*)d_in[0];
  const float* Fm    = (const float*)d_in[1];
  const float* Hm    = (const float*)d_in[2];
  const float* W_in  = (const float*)d_in[3];
  const float* b_in  = (const float*)d_in[4];
  const float* W_ih  = (const float*)d_in[5];
  const float* W_hh  = (const float*)d_in[6];
  const float* b_ih  = (const float*)d_in[7];
  const float* b_hh  = (const float*)d_in[8];
  const float* W_out = (const float*)d_in[9];
  const float* b_out = (const float*)d_in[10];
  float* out = (float*)d_out;
  bf16_t* wW = (bf16_t*)d_ws;   // 2*768*256*2 = 786,432 bytes

  // 1) weight convert + WMMA B-fragment swizzle (re-done each call: d_ws is
  //    poisoned by the harness; deterministic).
  int total = 2 * G3 * HID;
  knet_prep<<<(total + NTHREADS - 1) / NTHREADS, NTHREADS, 0, stream>>>(
      W_ih, W_hh, wW);

  // 2) persistent recurrent kernel, 184KB dynamic LDS per WG (CDNA5 320KB/WGP)
  (void)hipFuncSetAttribute((const void*)knet_main,
                            hipFuncAttributeMaxDynamicSharedMemorySize,
                            SMEM_TOTAL);
  knet_main<<<dim3(BATCHN / BT), dim3(NTHREADS), SMEM_TOTAL, stream>>>(
      y_seq, Fm, Hm, W_in, b_in, b_ih, b_hh, W_out, b_out, wW, out);
}
